// TranceptionAttention_1571958030986
// MI455X (gfx1250) — matrix operand order repaired and checked by reading, not compiled, not run on hardware
//
#include <hip/hip_runtime.h>
#include <hip/hip_bf16.h>

// Tranception "attention" — live graph after dead-code elimination:
//   qv = x @ [Wq[:,256:1024] | Wv[:,0:256]]        (GEMM1: 4096x1024x1024)
//   vm = [qv[:,768:1024],                                    heads 0-3
//         dwconv(dwconv(qv[:,  0:256], qw3)+qb3, vw3)+vb3,   heads 4-7
//         dwconv(dwconv(qv[:,256:512], qw5)+qb5, vw5)+vb5,   heads 8-11
//         dwconv(dwconv(qv[:,512:768], qw7)+qb7, vw7)+vb7]   heads 12-15
//   out = vm @ Wo + bo                             (GEMM2: 4096x1024x1024)
// GEMMs: bf16 v_wmma_f32_16x16x32_bf16, f32 accumulation, LDS double-buffered.
// Tile staging uses global_load_async_to_lds_b128 (ASYNCcnt): no VGPR data
// movement -> no spills, and the DMA overlaps the WMMA phase.

typedef unsigned short u16;
typedef __bf16 bf16_t;
typedef __attribute__((ext_vector_type(16))) bf16_t v16bf;
typedef __attribute__((ext_vector_type(8)))  float  v8f;
typedef int v4i __attribute__((vector_size(16)));

struct __align__(16) U4 { unsigned x, y, z, w; };
union FragBF { v16bf v; U4 u[2]; };

__device__ __forceinline__ u16 f2bf(float f) {
  union { float f; unsigned u; } c; c.f = f;
  unsigned r = c.u + 0x7FFFu + ((c.u >> 16) & 1u);   // round-to-nearest-even
  return (u16)(r >> 16);
}

// ---------------------------------------------------------------- prep ----
__global__ __launch_bounds__(256) void cvt_bf16(const float* __restrict__ in,
                                                u16* __restrict__ out, int n) {
  int i = blockIdx.x * 256 + threadIdx.x;
  if (i < n) out[i] = f2bf(in[i]);
}

// WqvT[n][k]: n<768 -> Wq[k][256+n] (q heads 4..15); n>=768 -> Wv[k][n-768] (v heads 0..3)
__global__ __launch_bounds__(256) void build_wqvT(const float* __restrict__ Wq,
                                                  const float* __restrict__ Wv,
                                                  u16* __restrict__ o) {
  int idx = blockIdx.x * 256 + threadIdx.x;        // 1024*1024 threads
  int n = idx >> 10, k = idx & 1023;
  float v = (n < 768) ? Wq[((size_t)k << 10) + 256 + n]
                      : Wv[((size_t)k << 10) + (n - 768)];
  o[idx] = f2bf(v);
}

__global__ __launch_bounds__(256) void build_T(const float* __restrict__ W,
                                               u16* __restrict__ o) {
  int idx = blockIdx.x * 256 + threadIdx.x;        // 1024*1024 threads
  int n = idx >> 10, k = idx & 1023;
  o[idx] = f2bf(W[((size_t)k << 10) + n]);
}

// ---------------------------------------------------------------- GEMM ----
// C[M,N] (f32, + optional bias) = A[M,K](bf16 row-major) * Bt[N,K](bf16)^T
// Block: 256 threads = 8 waves (2 M-waves x 4 N-waves); block tile 64M x 256N.
// Wave tile 32M x 64N: 2 A frags x 4 B frags x 8 accumulators per K=32 step.
// LDS rows padded 64B->80B so fragment ds_load_b128 reads spread over all banks.
// Fragment layouts per CDNA5 ISA 7.12.2/7.12.4:
//   A 16x32: lane<16 holds K{0..7,16..23}; lane>=16 holds K{8..15,24..31}
//   B 32x16: lane<16 holds K{0..7,16..23}; lane>=16 holds K{8..15,24..31}
//   D: VGPR r <-> M = r + 8*(lane>=16); lane&15 <-> N
#define LDS_ROW   40                          // u16 per LDS row (32 data + 8 pad)
#define LDS_ASZ   (64 * LDS_ROW)              //  2560 u16: A tile 64x32
#define LDS_BUFSZ (LDS_ASZ + 256 * LDS_ROW)   // 12800 u16: A + B tile 256x32

#if __has_builtin(__builtin_amdgcn_global_load_async_to_lds_b128)
#define HAVE_ASYNC_LDS 1
#if __has_builtin(__builtin_amdgcn_s_wait_asynccnt)
#define WAIT_ASYNC() __builtin_amdgcn_s_wait_asynccnt(0)
#else
#define WAIT_ASYNC() asm volatile("s_wait_asynccnt 0" ::: "memory")
#endif
__device__ __forceinline__ void async_cp16(const u16* g, u16* l) {
  __builtin_amdgcn_global_load_async_to_lds_b128(
      (__attribute__((address_space(1))) v4i*)(u16*)g,
      (__attribute__((address_space(3))) v4i*)l, 0, 0);
}
#else
#define HAVE_ASYNC_LDS 0
#define WAIT_ASYNC()
#endif

// stage one 64x32 A slab + 256x32 B slab (k = element offset) into LDS at sa
__device__ __forceinline__ void stage_tile(const u16* ag, const u16* bg, int k,
                                           u16* sa, int srow, int sc, int Kd) {
#if HAVE_ASYNC_LDS
  async_cp16(ag + k, sa + srow * LDS_ROW + sc);
#pragma unroll
  for (int j = 0; j < 4; ++j)
    async_cp16(bg + (size_t)j * 64 * Kd + k,
               sa + LDS_ASZ + (srow + j * 64) * LDS_ROW + sc);
#else
  U4 pa = *(const U4*)(ag + k);
  U4 pb[4];
#pragma unroll
  for (int j = 0; j < 4; ++j) pb[j] = *(const U4*)(bg + (size_t)j * 64 * Kd + k);
  *(U4*)(sa + srow * LDS_ROW + sc) = pa;
#pragma unroll
  for (int j = 0; j < 4; ++j)
    *(U4*)(sa + LDS_ASZ + (srow + j * 64) * LDS_ROW + sc) = pb[j];
#endif
}

// one K=32 step: 2 A frags x 4 B frags -> 8 WMMAs
__device__ __forceinline__ void gemm_step(const u16* la, int wm, int wn,
                                          int half, int r, v8f (&acc)[2][4]) {
  const u16* lb = la + LDS_ASZ;
  FragBF a0, a1;
  const u16* p0 = la + (wm * 32 + r) * LDS_ROW + half * 8;
  a0.u[0] = *(const U4*)(p0);
  a0.u[1] = *(const U4*)(p0 + 16);
  const u16* p1 = p0 + 16 * LDS_ROW;
  a1.u[0] = *(const U4*)(p1);
  a1.u[1] = *(const U4*)(p1 + 16);
#pragma unroll
  for (int t = 0; t < 4; ++t) {
    FragBF b;
    const u16* pbp = lb + (wn * 64 + t * 16 + r) * LDS_ROW + half * 8;
    b.u[0] = *(const U4*)(pbp);
    b.u[1] = *(const U4*)(pbp + 16);
    acc[0][t] = __builtin_amdgcn_wmma_f32_16x16x32_bf16(
        false, a0.v, false, b.v, (short)0, acc[0][t], false, false);
    acc[1][t] = __builtin_amdgcn_wmma_f32_16x16x32_bf16(
        false, a1.v, false, b.v, (short)0, acc[1][t], false, false);
  }
}

__global__ __launch_bounds__(256, 1) void gemm_bf16_wmma(
    const u16* __restrict__ A, const u16* __restrict__ Bt,
    const float* __restrict__ bias, float* __restrict__ C,
    int M, int N, int Kd) {
  __shared__ u16 lds[2 * LDS_BUFSZ];     // 51200 bytes

  const int tid  = threadIdx.x;
  const int lane = tid & 31;
  const int wave = tid >> 5;
  const int half = lane >> 4;
  const int r    = lane & 15;
  const int wm   = wave >> 2;            // 0..1  (M wave)
  const int wn   = wave & 3;             // 0..3  (N wave)
  const int m0   = blockIdx.x * 64;
  const int n0   = blockIdx.y * 256;

  // staging: each thread moves one 16B chunk of A and four of B per K-step
  const int srow = tid >> 2;             // 0..63
  const int sc   = (tid & 3) * 8;        // element offset within 32-elem row
  const u16* ag = A  + (size_t)(m0 + srow) * Kd + sc;
  const u16* bg = Bt + (size_t)(n0 + srow) * Kd + sc;

  u16* b0 = lds;
  u16* b1 = lds + LDS_BUFSZ;

  v8f acc[2][4] = {};
  const int nk = Kd / 32;                // even (Kd = 1024 -> 32)

  stage_tile(ag, bg, 0, b0, srow, sc, Kd);
  WAIT_ASYNC();
  __syncthreads();

  for (int kb = 0; kb < nk; kb += 2) {
    // loads for tile kb+1 land in b1 while the 8 WMMAs below consume b0;
    // all reads of b1 finished before the previous iteration's last barrier.
    if (kb + 1 < nk) stage_tile(ag, bg, (kb + 1) * 32, b1, srow, sc, Kd);
    gemm_step(b0, wm, wn, half, r, acc);
    WAIT_ASYNC();
    __syncthreads();

    if (kb + 1 < nk) {
      if (kb + 2 < nk) stage_tile(ag, bg, (kb + 2) * 32, b0, srow, sc, Kd);
      gemm_step(b1, wm, wn, half, r, acc);
      WAIT_ASYNC();
      __syncthreads();
    }
  }

  // epilogue
  const int mbase = m0 + wm * 32;
  const int nbase = n0 + wn * 64;
#pragma unroll
  for (int mi = 0; mi < 2; ++mi) {
#pragma unroll
    for (int t = 0; t < 4; ++t) {
      int ncol = nbase + t * 16 + r;
      float bv = (bias != nullptr) ? bias[ncol] : 0.0f;
      float* cp = C + (size_t)(mbase + mi * 16 + half * 8) * N + ncol;
#pragma unroll
      for (int vr = 0; vr < 8; ++vr)
        cp[(size_t)vr * N] = acc[mi][t][vr] + bv;
    }
  }
}

// ------------------------------------------------------- depthwise conv ---
// One thread per vm element (4096 rows x 1024 cols). Causal double conv along L.
__global__ __launch_bounds__(256) void dwconv_pack(
    const float* __restrict__ qv,
    const float* __restrict__ qw3, const float* __restrict__ qb3,
    const float* __restrict__ vw3, const float* __restrict__ vb3,
    const float* __restrict__ qw5, const float* __restrict__ qb5,
    const float* __restrict__ vw5, const float* __restrict__ vb5,
    const float* __restrict__ qw7, const float* __restrict__ qb7,
    const float* __restrict__ vw7, const float* __restrict__ vb7,
    u16* __restrict__ vm) {
  int idx = blockIdx.x * 256 + threadIdx.x;        // exactly 4096*1024 threads
  int row = idx >> 10;
  int c   = idx & 1023;
  float outv;
  if (c < 256) {
    // heads 0..3 = v group 0 = qv columns 768..1023
    outv = qv[((size_t)row << 10) + 768 + c];
  } else {
    int g  = c >> 8;                 // 1,2,3
    int ks = 2 * g + 1;              // 3,5,7
    int sc = ((g - 1) << 8) + (c & 255);  // source column in qv
    int d  = c & 63;                 // depthwise channel
    const float *qw, *qb, *vw, *vb;
    if (g == 1)      { qw = qw3; qb = qb3; vw = vw3; vb = vb3; }
    else if (g == 2) { qw = qw5; qb = qb5; vw = vw5; vb = vb5; }
    else             { qw = qw7; qb = qb7; vw = vw7; vb = vb7; }
    int l = row & 2047;              // position within sequence
    const float* src = qv + (((size_t)(row - l)) << 10) + sc;   // l = 0 row
    float acc = vb[d];
    for (int i = 0; i < ks; ++i) {   // second conv taps over y1
      int l1 = l - (ks - 1) + i;
      if (l1 < 0) continue;          // zero padding of y1
      float y1 = qb[d];
      for (int j = 0; j < ks; ++j) { // first conv taps over qv
        int t = l1 - (ks - 1) + j;
        if (t >= 0) y1 += qw[d * ks + j] * src[(size_t)t << 10];
      }
      acc += vw[d * ks + i] * y1;
    }
    outv = acc;
  }
  vm[idx] = f2bf(outv);
}

// ---------------------------------------------------------------- host ----
extern "C" void kernel_launch(void* const* d_in, const int* in_sizes, int n_in,
                              void* d_out, int out_size, void* d_ws, size_t ws_size,
                              hipStream_t stream) {
  (void)in_sizes; (void)n_in; (void)out_size; (void)ws_size;
  const float* x   = (const float*)d_in[0];
  const float* Wq  = (const float*)d_in[3];
  const float* Wv  = (const float*)d_in[5];
  const float* Wo  = (const float*)d_in[6];
  const float* bo  = (const float*)d_in[7];
  const float* qw3 = (const float*)d_in[8];
  const float* qb3 = (const float*)d_in[9];
  const float* vw3 = (const float*)d_in[12];
  const float* vb3 = (const float*)d_in[13];
  const float* qw5 = (const float*)d_in[14];
  const float* qb5 = (const float*)d_in[15];
  const float* vw5 = (const float*)d_in[18];
  const float* vb5 = (const float*)d_in[19];
  const float* qw7 = (const float*)d_in[20];
  const float* qb7 = (const float*)d_in[21];
  const float* vw7 = (const float*)d_in[24];
  const float* vb7 = (const float*)d_in[25];

  const int M = 4096, N = 1024, Kd = 1024;   // M = B*L
  char* ws = (char*)d_ws;
  u16*   xb   = (u16*)  (ws);                     //  8 MB  x in bf16
  u16*   WqvT = (u16*)  (ws + (size_t)( 8u<<20)); //  2 MB  fused [Wq'|Wv'] ^T bf16
  u16*   WoT  = (u16*)  (ws + (size_t)(10u<<20)); //  2 MB  Wo^T bf16
  float* qv   = (float*)(ws + (size_t)(12u<<20)); // 16 MB  GEMM1 output f32
  u16*   vm   = (u16*)  (ws + (size_t)(28u<<20)); //  8 MB  conv output bf16
  // total workspace: 36 MB

  cvt_bf16  <<<(M * Kd) / 256, 256, 0, stream>>>(x, xb, M * Kd);
  build_wqvT<<<(N * Kd) / 256, 256, 0, stream>>>(Wq, Wv, WqvT);
  build_T   <<<(N * Kd) / 256, 256, 0, stream>>>(Wo, WoT);

  dim3 gg(M / 64, N / 256);
  gemm_bf16_wmma<<<gg, 256, 0, stream>>>(xb, WqvT, nullptr, qv, M, N, Kd);

  dwconv_pack<<<(M * N) / 256, 256, 0, stream>>>(
      qv, qw3, qb3, vw3, vb3, qw5, qb5, vw5, vb5, qw7, qb7, vw7, vb7, vm);

  gemm_bf16_wmma<<<gg, 256, 0, stream>>>(vm, WoT, bo, (float*)d_out, M, N, Kd);
}
